// DiffusionTransformerBlock_78228534329669
// MI455X (gfx1250) — matrix operand order repaired and checked by reading, not compiled, not run on hardware
//
#include <hip/hip_runtime.h>
#include <cstdint>
#include <cstddef>

#define B_SZ 2
#define LEN  1024
#define CH   512
#define NH   16
#define HD   32
#define CZ   64
#define FFH  2048

typedef __attribute__((ext_vector_type(16))) __bf16 v16bf;
typedef __attribute__((ext_vector_type(8)))  float  v8f;

// ---------- helpers ----------
__device__ __forceinline__ unsigned short f2bf(float f) {
  union { float f; unsigned int u; } c; c.f = f;
  unsigned int r = (c.u + 0x7FFFu + ((c.u >> 16) & 1u)) >> 16;
  return (unsigned short)r;
}

// A-matrix (16x32 bf16) fragment: lane m = lane%16 is the row.
// lanes 0-15 hold K = {0..7, 16..23}; lanes 16-31 hold K = {8..15, 24..31}
__device__ __forceinline__ v16bf load_frag_a(const unsigned short* base, int hi) {
  union { v16bf v; float4 f4[2]; } u;
  u.f4[0] = *(const float4*)(base + hi * 8);
  u.f4[1] = *(const float4*)(base + 16 + hi * 8);
  return u.v;
}
// B-matrix (32x16 bf16) fragment: lane n = lane%16 is the column,
// lanes 0-15 hold K = {0..7, 16..23}; lanes 16-31 hold K = {8..15, 24..31}
__device__ __forceinline__ v16bf load_frag_b(const unsigned short* base, int hi) {
  union { v16bf v; float4 f4[2]; } u;
  u.f4[0] = *(const float4*)(base + hi * 8);
  u.f4[1] = *(const float4*)(base + 16 + hi * 8);
  return u.v;
}
__device__ __forceinline__ v8f wmma_bf16(v16bf a, v16bf b, v8f c) {
  return __builtin_amdgcn_wmma_f32_16x16x32_bf16(false, a, false, b, (short)0, c,
                                                 false, false);
}

// ---------- CDNA5 async memory->LDS (ASYNCcnt path, ISA 08_async_tensor) ----------
__device__ __forceinline__ void async_load_b128(unsigned lds_addr, const float* g) {
  asm volatile("global_load_async_to_lds_b128 %0, %1, off"
               :: "v"(lds_addr), "v"((unsigned long long)(uintptr_t)g)
               : "memory");
}
template <int N>
__device__ __forceinline__ void wait_asynccnt() {
  asm volatile("s_wait_asynccnt %0" :: "i"(N) : "memory");
}

// ---------- f32 -> bf16 weight conversion ----------
__global__ __launch_bounds__(256)
void cvt_bf16_kernel(const float* __restrict__ src, unsigned short* __restrict__ dst, int n) {
  int i = blockIdx.x * 256 + threadIdx.x;
  if (i < n) dst[i] = f2bf(src[i]);
}

// ---------- adaLN cond projection: ss[b, 0..2C) = time_cond[b] @ W.T + bias ----------
__global__ __launch_bounds__(256)
void cond_proj_kernel(const float* __restrict__ tc, const float* __restrict__ w,
                      const float* __restrict__ bias, float* __restrict__ ss) {
  int idx = blockIdx.x * 256 + threadIdx.x;     // 0 .. B*2C-1 (2048)
  int b = idx >> 10;
  int o = idx & 1023;
  const float4* tcv = (const float4*)(tc + (size_t)b * CH);
  const float4* wv  = (const float4*)(w  + (size_t)o * CH);
  float acc = bias[o];
  #pragma unroll 4
  for (int c = 0; c < CH / 4; ++c) {
    float4 a = tcv[c], q = wv[c];
    acc += a.x * q.x + a.y * q.y + a.z * q.z + a.w * q.w;
  }
  ss[idx] = acc;
}

// ---------- adaLN apply: out_bf16 = LN(x)*(1+scale) + shift ----------
__global__ __launch_bounds__(256)
void adaln_kernel(const float* __restrict__ x, const float* __restrict__ ss,
                  const float* __restrict__ g, const float* __restrict__ bta,
                  unsigned short* __restrict__ out) {
  __shared__ float red[16];
  int row = blockIdx.x;                 // b*L + l
  int b = row >> 10;
  int t = threadIdx.x;
  int lane = t & 31, wave = t >> 5;
  const float* xr = x + (size_t)row * CH;
  int c0 = t * 2;
  float v0 = xr[c0], v1 = xr[c0 + 1];
  float s = v0 + v1, q = v0 * v0 + v1 * v1;
  #pragma unroll
  for (int o = 16; o > 0; o >>= 1) { s += __shfl_xor(s, o, 32); q += __shfl_xor(q, o, 32); }
  if (lane == 0) { red[wave] = s; red[8 + wave] = q; }
  __syncthreads();
  float ts = 0.f, tq = 0.f;
  #pragma unroll
  for (int w = 0; w < 8; ++w) { ts += red[w]; tq += red[8 + w]; }
  float mu = ts * (1.f / CH);
  float var = tq * (1.f / CH) - mu * mu;
  float rs = rsqrtf(var + 1e-5f);
  const float* ssr = ss + (size_t)b * (2 * CH);
  #pragma unroll
  for (int e = 0; e < 2; ++e) {
    int c = c0 + e;
    float ln = (xr[c] - mu) * rs * g[c] + bta[c];
    float val = ln * (1.f + ssr[c]) + ssr[CH + c];
    out[(size_t)row * CH + c] = f2bf(val);
  }
}

// ---------- generic bf16 WMMA GEMM: out[M,N] = A[M,K] @ W[N,K]^T (+epilogue) ----------
enum { EP_QKV = 0, EP_V = 1, EP_RESID = 2, EP_GELU = 3 };

template <int EP>
__global__ __launch_bounds__(128)
void gemm_bf16_kernel(const unsigned short* __restrict__ A,
                      const unsigned short* __restrict__ W,
                      const float* __restrict__ bias,
                      const float* __restrict__ resid,
                      void* __restrict__ outp, int M, int N, int K) {
  int lane = threadIdx.x & 31, wave = threadIdx.x >> 5;
  int hi = lane >> 4, ln = lane & 15;
  int m0 = blockIdx.y * 16;
  int n0 = (blockIdx.x * 4 + wave) * 64;
  v8f acc[4];
  #pragma unroll
  for (int sIdx = 0; sIdx < 4; ++sIdx)
    #pragma unroll
    for (int r = 0; r < 8; ++r) acc[sIdx][r] = 0.0f;

  const unsigned short* arow = A + (size_t)(m0 + ln) * K;
  for (int k0 = 0; k0 < K; k0 += 32) {
    if (k0 + 64 < K) __builtin_prefetch(arow + k0 + 64, 0, 1);  // global_prefetch_b8
    v16bf af = load_frag_a(arow + k0, hi);
    #pragma unroll
    for (int sIdx = 0; sIdx < 4; ++sIdx) {
      const unsigned short* wrow = W + (size_t)(n0 + sIdx * 16 + ln) * K + k0;
      v16bf bfr = load_frag_b(wrow, hi);
      acc[sIdx] = wmma_bf16(af, bfr, acc[sIdx]);
    }
  }
  #pragma unroll
  for (int r = 0; r < 8; ++r) {
    int m = m0 + r + hi * 8;
    int bb = m >> 10, i = m & 1023;
    #pragma unroll
    for (int sIdx = 0; sIdx < 4; ++sIdx) {
      int n = n0 + sIdx * 16 + ln;
      float v = acc[sIdx][r];
      if (EP == EP_QKV) {                 // [B,H,L,D] bf16
        int h = n >> 5, d = n & 31;
        ((unsigned short*)outp)[(((size_t)bb * NH + h) * LEN + i) * HD + d] = f2bf(v);
      } else if (EP == EP_V) {            // transposed [B,H,D,L] bf16
        int h = n >> 5, d = n & 31;
        ((unsigned short*)outp)[(((size_t)bb * NH + h) * HD + d) * LEN + i] = f2bf(v);
      } else if (EP == EP_RESID) {        // f32: resid + acc + bias
        size_t idx = (size_t)m * N + n;
        ((float*)outp)[idx] = resid[idx] + v + bias[n];
      } else {                            // EP_GELU -> bf16
        float tv = v + bias[n];
        float gl = 0.5f * tv * (1.0f + erff(tv * 0.70710678118654752f));
        ((unsigned short*)outp)[(size_t)m * N + n] = f2bf(gl);
      }
    }
  }
}

// ---------- pair bias with async double-buffered LDS streaming ----------
// bias[b,h,i,j] = sum_c pair[b,i,j,c] * w_pair[h,c]
// One block per (b,i) row: 16 stages of 64 j x 64 c (16KB), global->LDS via
// global_load_async_to_lds_b128, next stage in flight while current reduces.
#define PB_J 64

__device__ __forceinline__ void pb_issue_stage(const float* __restrict__ chunk,
                                               unsigned lds_base, int t) {
  #pragma unroll
  for (int r = 0; r < 4; ++r) {
    int l4 = r * 256 + t;                 // float4 index within 64x64 tile
    int j  = l4 >> 4;                     // row (j) = l4 / 16
    int c4 = l4 & 15;                     // float4 column within row
    int sw = (l4 & ~15) | (c4 ^ (j & 15));  // XOR swizzle (16B units) vs bank conflicts
    async_load_b128(lds_base + (unsigned)sw * 16u, chunk + (size_t)l4 * 4);
  }
}

__global__ __launch_bounds__(256)
void pairbias_async_kernel(const float* __restrict__ pair, const float* __restrict__ wpair,
                           float* __restrict__ biasb) {
  __shared__ float stage[2][PB_J * CZ];   // 2 x 16KB
  __shared__ float wp[NH * CZ];           // 4KB
  int t = threadIdx.x;
  int i = blockIdx.x, b = blockIdx.y;

  ((float4*)wp)[t] = ((const float4*)wpair)[t];   // 1024 floats

  const float* prow = pair + ((size_t)b * LEN + i) * LEN * CZ;
  unsigned lds_base[2];
  lds_base[0] = (unsigned)(uintptr_t)(void*)&stage[0][0];  // low 32 bits = LDS offset
  lds_base[1] = (unsigned)(uintptr_t)(void*)&stage[1][0];

  int j  = t & 63;           // this thread's j within the chunk
  int hg = t >> 6;           // head group: computes heads hg*4 .. hg*4+3
  const float* wpr = wp + hg * 4 * CZ;
  float* brow0 = biasb + (((size_t)b * NH + hg * 4) * LEN + i) * LEN;

  pb_issue_stage(prow, lds_base[0], t);

  #pragma unroll 1
  for (int c = 0; c < LEN / PB_J; ++c) {
    if (c + 1 < LEN / PB_J) {
      pb_issue_stage(prow + (size_t)(c + 1) * PB_J * CZ, lds_base[(c + 1) & 1], t);
      wait_asynccnt<4>();     // this wave's current stage (first 4) complete
    } else {
      wait_asynccnt<0>();
    }
    __syncthreads();

    const float* st = &stage[c & 1][0];
    float acc0 = 0.f, acc1 = 0.f, acc2 = 0.f, acc3 = 0.f;
    #pragma unroll 4
    for (int cc = 0; cc < CZ; ++cc) {
      int c4 = cc >> 2;
      float val = st[j * CZ + ((c4 ^ (j & 15)) << 2) + (cc & 3)];
      acc0 += val * wpr[0 * CZ + cc];
      acc1 += val * wpr[1 * CZ + cc];
      acc2 += val * wpr[2 * CZ + cc];
      acc3 += val * wpr[3 * CZ + cc];
    }
    size_t jo = (size_t)c * PB_J + j;
    brow0[0 * (size_t)LEN * LEN + jo] = acc0;
    brow0[1 * (size_t)LEN * LEN + jo] = acc1;
    brow0[2 * (size_t)LEN * LEN + jo] = acc2;
    brow0[3 * (size_t)LEN * LEN + jo] = acc3;
    __syncthreads();
  }
}

// ---------- attention: per (b, head, 16-row i-tile), S row-block in LDS ----------
__global__ __launch_bounds__(64)
void attn_kernel(const unsigned short* __restrict__ q, const unsigned short* __restrict__ k,
                 const unsigned short* __restrict__ v, const float* __restrict__ biasb,
                 unsigned short* __restrict__ attn_out) {
  __shared__ float S[16][LEN];   // exactly 64KB
  int lane = threadIdx.x & 31, wave = threadIdx.x >> 5;
  int hi = lane >> 4, ln = lane & 15;
  int i0 = blockIdx.x * 16, head = blockIdx.y, b = blockIdx.z;
  int bh = b * NH + head;

  const unsigned short* qbase = q + ((size_t)bh * LEN + i0) * HD;
  v16bf qf = load_frag_a(qbase + ln * HD, hi);   // K = full HEAD_DIM = 32

  // phase 1: S = Q K^T  (each wave handles alternating j tiles)
  for (int jt = wave; jt < LEN / 16; jt += 2) {
    int j0 = jt * 16;
    const unsigned short* kbase = k + ((size_t)bh * LEN + j0 + ln) * HD;
    v16bf kf = load_frag_b(kbase, hi);
    v8f sacc;
    #pragma unroll
    for (int r = 0; r < 8; ++r) sacc[r] = 0.0f;
    sacc = wmma_bf16(qf, kf, sacc);
    #pragma unroll
    for (int r = 0; r < 8; ++r) S[r + hi * 8][j0 + ln] = sacc[r];
  }
  __syncthreads();

  // phase 2: scale + pair bias + softmax, normalized P left in LDS (f32)
  const float scale = 0.17677669529663687f;   // 1/sqrt(32)
  for (int rr = 0; rr < 8; ++rr) {
    int r = wave * 8 + rr;
    const float* brow = biasb + ((size_t)bh * LEN + i0 + r) * LEN;
    float mx = -1e30f;
    for (int jj = lane; jj < LEN; jj += 32) {
      float vv = S[r][jj] * scale + brow[jj];
      S[r][jj] = vv;
      mx = fmaxf(mx, vv);
    }
    #pragma unroll
    for (int o = 16; o > 0; o >>= 1) mx = fmaxf(mx, __shfl_xor(mx, o, 32));
    float sm = 0.f;
    for (int jj = lane; jj < LEN; jj += 32) {
      float e = __expf(S[r][jj] - mx);
      S[r][jj] = e;
      sm += e;
    }
    #pragma unroll
    for (int o = 16; o > 0; o >>= 1) sm += __shfl_xor(sm, o, 32);
    float inv = 1.0f / sm;
    for (int jj = lane; jj < LEN; jj += 32) S[r][jj] *= inv;
  }
  __syncthreads();

  // phase 3: O = P @ V  (wave 0 -> d 0..15, wave 1 -> d 16..31)
  int n0 = wave * 16;
  v8f oacc;
  #pragma unroll
  for (int r = 0; r < 8; ++r) oacc[r] = 0.0f;
  for (int jt = 0; jt < LEN / 16; ++jt) {
    int j0 = jt * 16;
    union { v16bf vv; unsigned short us[16]; } pa;   // P fragment, f32->bf16 on the fly
    #pragma unroll
    for (int e = 0; e < 8; ++e) {
      pa.us[e]     = f2bf(S[ln][j0 + hi * 8 + e]);
      pa.us[e + 8] = f2bf(S[ln][j0 + 16 + hi * 8 + e]);
    }
    const unsigned short* vbase = v + ((size_t)bh * HD + n0 + ln) * LEN + j0;
    v16bf vf = load_frag_b(vbase, hi);
    oacc = wmma_bf16(pa.vv, vf, oacc);
  }
  #pragma unroll
  for (int r = 0; r < 8; ++r) {
    int m = r + hi * 8;
    int n = n0 + ln;
    attn_out[((size_t)b * LEN + i0 + m) * CH + head * HD + n] = f2bf(oacc[r]);
  }
}

// ---------- host-side orchestration ----------
extern "C" void kernel_launch(void* const* d_in, const int* in_sizes, int n_in,
                              void* d_out, int out_size, void* d_ws, size_t ws_size,
                              hipStream_t stream) {
  (void)in_sizes; (void)n_in; (void)out_size; (void)ws_size;
  const float* x       = (const float*)d_in[0];
  const float* pair    = (const float*)d_in[1];
  const float* tc      = (const float*)d_in[2];
  const float* ln1_g   = (const float*)d_in[3];
  const float* ln1_b   = (const float*)d_in[4];
  const float* ada1_w  = (const float*)d_in[5];
  const float* ada1_b  = (const float*)d_in[6];
  const float* wq      = (const float*)d_in[7];
  const float* wk      = (const float*)d_in[8];
  const float* wv      = (const float*)d_in[9];
  const float* w_pair  = (const float*)d_in[10];
  const float* wo      = (const float*)d_in[11];
  const float* bo      = (const float*)d_in[12];
  const float* ln2_g   = (const float*)d_in[13];
  const float* ln2_b   = (const float*)d_in[14];
  const float* ada2_w  = (const float*)d_in[15];
  const float* ada2_b  = (const float*)d_in[16];
  const float* w1      = (const float*)d_in[17];
  const float* b1      = (const float*)d_in[18];
  const float* w2      = (const float*)d_in[19];
  const float* b2      = (const float*)d_in[20];

  char* wsb = (char*)d_ws;
  size_t off = 0;
  auto take = [&](size_t bytes) -> void* {
    void* p = wsb + off;
    off += (bytes + 255) & ~(size_t)255;
    return p;
  };
  const size_t M = (size_t)B_SZ * LEN;                       // 2048 rows
  float*          ss1     = (float*)take(B_SZ * 2 * CH * 4);
  float*          ss2     = (float*)take(B_SZ * 2 * CH * 4);
  unsigned short* h_bf    = (unsigned short*)take(M * CH * 2);
  unsigned short* h2_bf   = (unsigned short*)take(M * CH * 2);
  unsigned short* q_bf    = (unsigned short*)take(M * CH * 2);
  unsigned short* k_bf    = (unsigned short*)take(M * CH * 2);
  unsigned short* v_bf    = (unsigned short*)take(M * CH * 2);
  unsigned short* attn_bf = (unsigned short*)take(M * CH * 2);
  float*          x1      = (float*)take(M * CH * 4);
  unsigned short* mid_bf  = (unsigned short*)take(M * FFH * 2);
  unsigned short* wq_bf   = (unsigned short*)take((size_t)CH * CH * 2);
  unsigned short* wk_bf   = (unsigned short*)take((size_t)CH * CH * 2);
  unsigned short* wv_bf   = (unsigned short*)take((size_t)CH * CH * 2);
  unsigned short* wo_bf   = (unsigned short*)take((size_t)CH * CH * 2);
  unsigned short* w1_bf   = (unsigned short*)take((size_t)FFH * CH * 2);
  unsigned short* w2_bf   = (unsigned short*)take((size_t)CH * FFH * 2);
  float*          biasb   = (float*)take((size_t)B_SZ * NH * LEN * LEN * 4);

  // 1) weight conversion f32 -> bf16
  int ncc = CH * CH;
  cvt_bf16_kernel<<<ncc / 256, 256, 0, stream>>>(wq, wq_bf, ncc);
  cvt_bf16_kernel<<<ncc / 256, 256, 0, stream>>>(wk, wk_bf, ncc);
  cvt_bf16_kernel<<<ncc / 256, 256, 0, stream>>>(wv, wv_bf, ncc);
  cvt_bf16_kernel<<<ncc / 256, 256, 0, stream>>>(wo, wo_bf, ncc);
  int nff = FFH * CH;
  cvt_bf16_kernel<<<nff / 256, 256, 0, stream>>>(w1, w1_bf, nff);
  cvt_bf16_kernel<<<nff / 256, 256, 0, stream>>>(w2, w2_bf, nff);

  // 2) adaLN cond projections
  cond_proj_kernel<<<(B_SZ * 2 * CH) / 256, 256, 0, stream>>>(tc, ada1_w, ada1_b, ss1);
  cond_proj_kernel<<<(B_SZ * 2 * CH) / 256, 256, 0, stream>>>(tc, ada2_w, ada2_b, ss2);

  // 3) adaLN #1 -> h (bf16)
  adaln_kernel<<<(int)M, 256, 0, stream>>>(x, ss1, ln1_g, ln1_b, h_bf);

  // 4) Q/K/V projections (WMMA bf16)
  dim3 gproj(CH / 256, (int)M / 16);
  gemm_bf16_kernel<EP_QKV><<<gproj, 128, 0, stream>>>(h_bf, wq_bf, nullptr, nullptr, q_bf, (int)M, CH, CH);
  gemm_bf16_kernel<EP_QKV><<<gproj, 128, 0, stream>>>(h_bf, wk_bf, nullptr, nullptr, k_bf, (int)M, CH, CH);
  gemm_bf16_kernel<EP_V  ><<<gproj, 128, 0, stream>>>(h_bf, wv_bf, nullptr, nullptr, v_bf, (int)M, CH, CH);

  // 5) pair bias: async double-buffered streaming of the 512MB pair tensor
  pairbias_async_kernel<<<dim3(LEN, B_SZ), 256, 0, stream>>>(pair, w_pair, biasb);

  // 6) attention (WMMA QK^T, softmax in LDS, WMMA PV)
  attn_kernel<<<dim3(LEN / 16, NH, B_SZ), 64, 0, stream>>>(q_bf, k_bf, v_bf, biasb, attn_bf);

  // 7) output projection + residual -> x1 (f32)
  gemm_bf16_kernel<EP_RESID><<<gproj, 128, 0, stream>>>(attn_bf, wo_bf, bo, x, x1, (int)M, CH, CH);

  // 8) adaLN #2 -> h2 (bf16)
  adaln_kernel<<<(int)M, 256, 0, stream>>>(x1, ss2, ln2_g, ln2_b, h2_bf);

  // 9) FFN1 + exact GELU -> mid (bf16)
  dim3 gff1(FFH / 256, (int)M / 16);
  gemm_bf16_kernel<EP_GELU><<<gff1, 128, 0, stream>>>(h2_bf, w1_bf, b1, nullptr, mid_bf, (int)M, FFH, CH);

  // 10) FFN2 + residual -> d_out (f32)
  gemm_bf16_kernel<EP_RESID><<<gproj, 128, 0, stream>>>(mid_bf, w2_bf, b2, x1, (float*)d_out, (int)M, CH, FFH);
}
